// SinkhornDivergence_90177133347125
// MI455X (gfx1250) — matrix operand order repaired and checked by reading, not compiled, not run on hardware
//
#include <hip/hip_runtime.h>

typedef __attribute__((ext_vector_type(16))) __bf16 v16bf;
typedef __attribute__((ext_vector_type(8)))  float  v8f;
typedef unsigned int u32x4 __attribute__((ext_vector_type(4)));
typedef int          i32x4 __attribute__((ext_vector_type(4)));
typedef int          i32x8 __attribute__((ext_vector_type(8)));

#define BB 8
#define LL 2048
#define DD 32
#define CHUNK 256                 // key columns staged per LDS chunk
#define NCH   (LL / CHUNK)
#define KROWB 80                  // padded LDS bytes per key column (64B data + 16B pad)

#define LOG2E  1.4426950408889634f
#define HLOG2E 0.7213475204444817f
#define LN2    0.6931471805599453f

union BV16 { uint4 u[2]; v16bf v; };

#if __has_builtin(__builtin_amdgcn_tensor_load_to_lds) && __has_builtin(__builtin_amdgcn_s_wait_tensorcnt)
#define SK_TDM 1
#if __has_include(<hip/amd_detail/amd_gfx1250_TDM.h>)
#define SK_TDM6 1                 // clang-23 lane: 6-arg builtin
#endif
#endif

// ---------------------------------------------------------------------------
// Stage one chunk of the key matrix (CHUNK cols x 32 bf16) into LDS with an
// 80B row stride. TDM path: wave 0 issues one 2D tensor DMA with pad_enable
// (pad 4 dwords after every 16 dwords); fallback: cooperative vector copy.
// ---------------------------------------------------------------------------
__device__ __forceinline__
void sk_stage_keys(const __bf16* __restrict__ kbb, int chunk, char* buf, int tidx)
{
#ifdef SK_TDM
    if ((tidx >> 5) == 0) {
        unsigned long long ga = (unsigned long long)(uintptr_t)kbb
                              + (unsigned long long)(chunk * CHUNK) * (DD * 2);
        unsigned glo = (unsigned)ga;
        unsigned ghi = (unsigned)(ga >> 32);
        unsigned lds = (unsigned)(uintptr_t)buf;
        // Group 0: count=1, lds_addr, global_addr[56:0], type=2
        u32x4 g0 = { 1u, lds, glo, (ghi & 0x01FFFFFFu) | 0x80000000u };
        // Group 1: data_size=3 (8B), pad_enable, pad_interval=16dw, pad_amount=4dw,
        //          tensor_dim0=8, tensor_dim1=2048, tile_dim0=8, tile_dim1=CHUNK,
        //          tensor_dim0_stride=8
        i32x8 g1 = { (int)((3u << 16) | (1u << 20) | (3u << 22) | (3u << 25)),
                     (int)(8u << 16),
                     (int)(2048u << 16),
                     (int)(8u << 16),
                     (int)CHUNK,
                     8, 0, 0 };
        i32x4 gz = { 0, 0, 0, 0 };
#ifdef SK_TDM6
        i32x8 gz8 = { 0, 0, 0, 0, 0, 0, 0, 0 };
        __builtin_amdgcn_tensor_load_to_lds(g0, g1, gz, gz, gz8, 0);
#else
        __builtin_amdgcn_tensor_load_to_lds(g0, g1, gz, gz, 0);
#endif
    }
#else
    const char* gsrc = (const char*)kbb + (size_t)(chunk * CHUNK) * (DD * 2);
    for (int p = tidx; p < CHUNK * 4; p += 128) {
        int col = p >> 2, pc = p & 3;
        uint4 v = *(const uint4*)(gsrc + col * 64 + pc * 16);
        *(uint4*)(buf + col * KROWB + pc * 16) = v;
    }
#endif
}

__device__ __forceinline__ void sk_wait_keys()
{
#ifdef SK_TDM
    if ((threadIdx.x >> 5) == 0) __builtin_amdgcn_s_wait_tensorcnt(0);
#endif
}

// ---------------------------------------------------------------------------
// Prep: bf16 copy of points, squared norms, log weights, zero potentials.
// ---------------------------------------------------------------------------
__global__ __launch_bounds__(256)
void sk_prep(const float* __restrict__ X, const float* __restrict__ w,
             __bf16* __restrict__ Xh, float* __restrict__ nrm,
             float* __restrict__ logw, float* __restrict__ z0,
             float* __restrict__ z1, int nrows)
{
    int r = blockIdx.x * blockDim.x + threadIdx.x;
    if (r >= nrows) return;
    const float* xr = X + (size_t)r * DD;
    __bf16* xo = Xh + (size_t)r * DD;
    float s = 0.f;
#pragma unroll
    for (int d = 0; d < DD; ++d) {
        float v = xr[d];
        s += v * v;
        xo[d] = (__bf16)v;
    }
    nrm[r]  = s;
    logw[r] = __logf(w[r]);
    z0[r] = 0.f;
    z1[r] = 0.f;
}

// ---------------------------------------------------------------------------
// Fused c_transform (exp2 domain): for each query row l,
//   fn[l] = -log sum_k exp( min(q_l.k_n - 0.5|q|^2 - 0.5|k|^2, 0) + pot[n]+logw[n] )
// One V_WMMA_F32_16X16X32_BF16 per 16x16 tile; keys streamed through LDS via
// TDM double buffering; B fragments software-pipelined in registers so the
// WMMA only takes a partial dscnt wait. navg==1: out = 0.5*(fn+prev).
// 4 waves/WG, 16 rows/wave, grid = (L/64, B).
// ---------------------------------------------------------------------------
__global__ __launch_bounds__(128)
void sk_ctrans(const __bf16* __restrict__ Q,  const float* __restrict__ nq,
               const __bf16* __restrict__ Kb, const float* __restrict__ nk,
               const float* __restrict__ pot, const float* __restrict__ logw,
               const float* __restrict__ prev, float* __restrict__ outp,
               int navg)
{
    __shared__ char   ksm[2][CHUNK * KROWB];   // 2 x 20 KB key chunks (padded rows)
    __shared__ float2 cs[LL];                  // 16 KB: { log2e*0.5*|k|^2, log2e*(pot+logw) }

    const int b    = blockIdx.y;
    const int lane = threadIdx.x & 31;
    const int half = lane >> 4;
    const int ln   = lane & 15;
    const int rowbase = blockIdx.x * 64 + (threadIdx.x >> 5) * 16;
    const size_t bq = (size_t)b * LL;

    const __bf16* kbb  = Kb   + bq * DD;
    const float*  nkb  = nk   + bq;
    const float*  potb = pot  + bq;
    const float*  lwb  = logw + bq;

    // kick off first key DMA immediately
    sk_stage_keys(kbb, 0, ksm[0], threadIdx.x);

    // stage column metadata (whole batch, once per block)
    for (int c = threadIdx.x; c < LL; c += 128)
        cs[c] = make_float2(HLOG2E * nkb[c], LOG2E * (potb[c] + lwb[c]));

    // A tile (16 rows x 32 d): lane holds row rowbase+ln,
    // d in {half*8 .. +7} U {16+half*8 .. +7}; prescale by log2(e)
    const __bf16* qp = Q + (bq + rowbase + ln) * DD + half * 8;
    BV16 aq;
    aq.u[0] = *(const uint4*)(qp);
    aq.u[1] = *(const uint4*)(qp + 16);
#pragma unroll
    for (int i = 0; i < 16; ++i)
        aq.v[i] = (__bf16)((float)aq.v[i] * LOG2E);

    // log2e * 0.5*|q|^2 for this lane's 8 C rows
    float nqh[8];
    const float* nqp = nq + bq + rowbase + half * 8;
#pragma unroll
    for (int i = 0; i < 8; ++i) nqh[i] = HLOG2E * nqp[i];

    float mr[8], lr[8];
#pragma unroll
    for (int i = 0; i < 8; ++i) { mr[i] = -3.0e38f; lr[i] = 0.f; }

    for (int c = 0; c < NCH; ++c) {
        sk_wait_keys();          // wave 0: DMA for chunk c complete
        __syncthreads();         // all waves: chunk c visible, chunk (c+1)&1 buffer free
        if (c + 1 < NCH)
            sk_stage_keys(kbb, c + 1, ksm[(c + 1) & 1], threadIdx.x);

        const char* kbuf = ksm[c & 1];

        // prologue: fetch tile 0 operands into registers
        BV16 bcur;
        {
            const char* kp = kbuf + ln * KROWB + half * 16;
            bcur.u[0] = *(const uint4*)(kp);
            bcur.u[1] = *(const uint4*)(kp + 32);
        }
        float2 czc = cs[c * CHUNK + ln];

#pragma unroll 2
        for (int t = 0; t < CHUNK; t += 16) {
            // prefetch next tile (clamped on last iteration) before consuming
            const int tn = (t + 16 < CHUNK) ? (t + 16) : t;
            BV16 bnxt;
            const char* kp = kbuf + (tn + ln) * KROWB + half * 16;
            bnxt.u[0] = *(const uint4*)(kp);
            bnxt.u[1] = *(const uint4*)(kp + 32);
            float2 czn = cs[c * CHUNK + tn + ln];

            v8f acc = {};
            acc = __builtin_amdgcn_wmma_f32_16x16x32_bf16(false, aq.v, false, bcur.v,
                                                          (short)0, acc, false, false);
#pragma unroll
            for (int i = 0; i < 8; ++i) {
                float v  = fminf(acc[i] - nqh[i] - czc.x, 0.0f) + czc.y;  // base-2 score
                float mn = fmaxf(mr[i], v);
                lr[i] = lr[i] * __builtin_amdgcn_exp2f(mr[i] - mn)
                      + __builtin_amdgcn_exp2f(v - mn);
                mr[i] = mn;
            }
            bcur = bnxt;
            czc  = czn;
        }
    }

    // merge (m,l) across the 16 lanes that share each row (xor stays in half-wave)
#pragma unroll
    for (int off = 1; off < 16; off <<= 1) {
#pragma unroll
        for (int i = 0; i < 8; ++i) {
            float mo = __shfl_xor(mr[i], off, 32);
            float lo = __shfl_xor(lr[i], off, 32);
            float mn = fmaxf(mr[i], mo);
            lr[i] = lr[i] * __builtin_amdgcn_exp2f(mr[i] - mn)
                  + lo    * __builtin_amdgcn_exp2f(mo - mn);
            mr[i] = mn;
        }
    }

    if (ln == 0) {
        float*       ob = outp + bq + rowbase + half * 8;
        const float* pb = prev + bq + rowbase + half * 8;
#pragma unroll
        for (int i = 0; i < 8; ++i) {
            float fn = -LN2 * (mr[i] + __builtin_amdgcn_logf(lr[i]));
            ob[i] = navg ? 0.5f * (fn + pb[i]) : fn;
        }
    }
}

// ---------------------------------------------------------------------------
// Per-batch weighted sum: partial[b] = sum_r w[b,r]*f[b,r]
// ---------------------------------------------------------------------------
__global__ __launch_bounds__(256)
void sk_wsum(const float* __restrict__ w, const float* __restrict__ f,
             float* __restrict__ partial)
{
    __shared__ float sm[256];
    const int b = blockIdx.x;
    const size_t base = (size_t)b * LL;
    float s = 0.f;
    for (int r = threadIdx.x; r < LL; r += 256)
        s += w[base + r] * f[base + r];
    sm[threadIdx.x] = s;
    __syncthreads();
    for (int st = 128; st > 0; st >>= 1) {
        if (threadIdx.x < st) sm[threadIdx.x] += sm[threadIdx.x + st];
        __syncthreads();
    }
    if (threadIdx.x == 0) partial[b] = sm[0];
}

__global__ void sk_final(const float* __restrict__ p, float* __restrict__ out)
{
    if (threadIdx.x == 0 && blockIdx.x == 0) {
        float s = 0.f;
        for (int b = 0; b < BB; ++b)
            s += p[b] + p[BB + b] - p[2 * BB + b] - p[3 * BB + b];
        out[0] = s * (1.0f / BB);
    }
}

// ---------------------------------------------------------------------------
extern "C" void kernel_launch(void* const* d_in, const int* in_sizes, int n_in,
                              void* d_out, int out_size, void* d_ws, size_t ws_size,
                              hipStream_t stream)
{
    (void)in_sizes; (void)n_in; (void)out_size; (void)ws_size;

    const float* x   = (const float*)d_in[0];   // [8,2048,32]
    const float* a   = (const float*)d_in[1];   // [8,2048]
    const float* y   = (const float*)d_in[2];   // [8,2048,32]
    const float* bwt = (const float*)d_in[3];   // [8,2048]

    const size_t NR = (size_t)BB * LL;
    char* base = (char*)d_ws;
    size_t off = 0;
    auto alloc = [&](size_t bytes) -> void* {
        void* p = base + off;
        off = (off + bytes + 255) & ~(size_t)255;
        return p;
    };
    __bf16* xh = (__bf16*)alloc(NR * DD * sizeof(__bf16));
    __bf16* yh = (__bf16*)alloc(NR * DD * sizeof(__bf16));
    float* nx  = (float*)alloc(NR * 4);
    float* ny  = (float*)alloc(NR * 4);
    float* la  = (float*)alloc(NR * 4);
    float* lb  = (float*)alloc(NR * 4);
    float* f0  = (float*)alloc(NR * 4);
    float* f1  = (float*)alloc(NR * 4);
    float* g0  = (float*)alloc(NR * 4);
    float* g1  = (float*)alloc(NR * 4);
    float* fx0 = (float*)alloc(NR * 4);
    float* fx1 = (float*)alloc(NR * 4);
    float* fy0 = (float*)alloc(NR * 4);
    float* fy1 = (float*)alloc(NR * 4);
    float* tb  = (float*)alloc(NR * 4);
    float* part = (float*)alloc(4 * BB * 4);

    dim3 pgrid((unsigned)((NR + 255) / 256)), pblk(256);
    sk_prep<<<pgrid, pblk, 0, stream>>>(x, a,   xh, nx, la, f0, fx0, (int)NR);
    sk_prep<<<pgrid, pblk, 0, stream>>>(y, bwt, yh, ny, lb, g0, fy0, (int)NR);

    dim3 cgrid(LL / 64, BB), cblk(128);

    // Sinkhorn xy iterations: fn from g, gn from f, both averaged.
    float *fc = f0, *fa = f1, *gc = g0, *ga = g1;
    for (int it = 0; it < 10; ++it) {
        sk_ctrans<<<cgrid, cblk, 0, stream>>>(xh, nx, yh, ny, gc, lb, fc, fa, 1);
        sk_ctrans<<<cgrid, cblk, 0, stream>>>(yh, ny, xh, nx, fc, la, gc, ga, 1);
        float* t;
        t = fc; fc = fa; fa = t;
        t = gc; gc = ga; ga = t;
    }

    // Symmetric potentials.
    float *xc = fx0, *xa = fx1, *yc = fy0, *ya = fy1;
    for (int it = 0; it < 10; ++it) {
        sk_ctrans<<<cgrid, cblk, 0, stream>>>(xh, nx, xh, nx, xc, la, xc, xa, 1);
        sk_ctrans<<<cgrid, cblk, 0, stream>>>(yh, ny, yh, ny, yc, lb, yc, ya, 1);
        float* t;
        t = xc; xc = xa; xa = t;
        t = yc; yc = ya; ya = t;
    }

    // Final evaluations: raw c_transform then deterministic weighted sums.
    sk_ctrans<<<cgrid, cblk, 0, stream>>>(xh, nx, yh, ny, gc, lb, la, tb, 0);
    sk_wsum<<<dim3(BB), dim3(256), 0, stream>>>(a, tb, part + 0);

    sk_ctrans<<<cgrid, cblk, 0, stream>>>(yh, ny, xh, nx, fc, la, lb, tb, 0);
    sk_wsum<<<dim3(BB), dim3(256), 0, stream>>>(bwt, tb, part + BB);

    sk_ctrans<<<cgrid, cblk, 0, stream>>>(xh, nx, xh, nx, xc, la, la, tb, 0);
    sk_wsum<<<dim3(BB), dim3(256), 0, stream>>>(a, tb, part + 2 * BB);

    sk_ctrans<<<cgrid, cblk, 0, stream>>>(yh, ny, yh, ny, yc, lb, lb, tb, 0);
    sk_wsum<<<dim3(BB), dim3(256), 0, stream>>>(bwt, tb, part + 3 * BB);

    sk_final<<<dim3(1), dim3(32), 0, stream>>>(part, (float*)d_out);
}
